// AdditiveAttention_77953656422782
// MI455X (gfx1250) — matrix operand order repaired and checked by reading, not compiled, not run on hardware
//
#include <hip/hip_runtime.h>
#include <hip/hip_bf16.h>

typedef __attribute__((ext_vector_type(4)))  _Float16 v4h;
typedef __attribute__((ext_vector_type(8)))  _Float16 v8h;
typedef __attribute__((ext_vector_type(16))) _Float16 v16h;
typedef __attribute__((ext_vector_type(4)))  float    v4f;
typedef __attribute__((ext_vector_type(8)))  float    v8f;

#define B_SZ 128
#define N_SZ 196
#define HDIM 512
#define VDIM 2048
#define ATT  512
#define ROWS (B_SZ * N_SZ)          // 25088 = 784 * 32

// LDS strides padded to 40 halves (80 bytes) so stride-across-lanes fragment
// reads (20 banks apart) touch 16 distinct bank groups.
#define A_STRIDE 40
#define B_STRIDE 40

// gfx1250 async global->LDS path (ASYNCcnt), guarded so absence falls back.
// Param types per hipcc diagnostic: (v4i AS1*, v4i AS3*, imm offset, imm cpol)
#if __has_builtin(__builtin_amdgcn_global_load_async_to_lds_b128) && \
    __has_builtin(__builtin_amdgcn_s_wait_asynccnt)
#define HAVE_ASYNC_LDS 1
typedef int v4i_t __attribute__((__vector_size__(4 * sizeof(int))));
typedef __attribute__((address_space(1))) v4i_t glob_v4i;
typedef __attribute__((address_space(3))) v4i_t lds_v4i;
#endif

// ---------------------------------------------------------------------------
// Fast device transcendentals: prefer HW tanh; else branch-free exp2/rcp form.
// tanh(x) = 1 - 2/(e^(2x)+1): exact at +-inf, NaN-propagating, no branches.
// ---------------------------------------------------------------------------
__device__ __forceinline__ float fast_tanh(float x) {
#if __has_builtin(__builtin_amdgcn_tanhf)
    return __builtin_amdgcn_tanhf(x);
#elif __has_builtin(__builtin_amdgcn_tanh_f32)
    return __builtin_amdgcn_tanh_f32(x);
#else
    const float e = __builtin_amdgcn_exp2f(x * 2.885390081777927f); // 2*log2(e)
    return 1.0f - 2.0f * __builtin_amdgcn_rcpf(e + 1.0f);
#endif
}

__device__ __forceinline__ float fast_exp(float x) {
    return __builtin_amdgcn_exp2f(x * 1.4426950408889634f);
}

// ---------------------------------------------------------------------------
// Kernel 1: transpose + convert Uw [VDIM,ATT] f32 -> UwT [ATT,VDIM] f16
// ---------------------------------------------------------------------------
__global__ void uw_transpose_f16(const float* __restrict__ Uw,
                                 _Float16* __restrict__ UwT) {
    __shared__ float tile[32][33];
    const int k0 = blockIdx.x * 32;   // VDIM
    const int a0 = blockIdx.y * 32;   // ATT
    const int tx = threadIdx.x & 31;
    const int ty = threadIdx.x >> 5;  // 0..7
#pragma unroll
    for (int i = 0; i < 32; i += 8)
        tile[ty + i][tx] = Uw[(size_t)(k0 + ty + i) * ATT + a0 + tx];
    __syncthreads();
#pragma unroll
    for (int i = 0; i < 32; i += 8)
        UwT[(size_t)(a0 + ty + i) * VDIM + (k0 + tx)] =
            (_Float16)tile[tx][ty + i];
}

// ---------------------------------------------------------------------------
// Kernel 2: whUb[b,a] = h[b,:] @ Ww[:,a] + Wb[a] + Ub[a]   (fp32, tiny)
// ---------------------------------------------------------------------------
__global__ void wh_kernel(const float* __restrict__ h,
                          const float* __restrict__ Ww,
                          const float* __restrict__ Wb,
                          const float* __restrict__ Ub,
                          float* __restrict__ whUb) {
    const int a = blockIdx.x * 256 + threadIdx.x;
    const int b = blockIdx.y;
    float acc = 0.f;
    for (int k = 0; k < HDIM; ++k)
        acc += h[b * HDIM + k] * Ww[(size_t)k * ATT + a];
    whUb[b * ATT + a] = acc + Wb[a] + Ub[a];
}

// ---------------------------------------------------------------------------
// Kernel 3: main WMMA kernel.
// escore[r] = vw . tanh( V[r,:] @ Uw + whUb[b(r),:] ) + vb, r = b*196+n
// Workgroup: 256 thr (8 waves), 32 rows x 512 cols, K chunked by 32 via LDS.
// Wave w: row subtile sm = w>>2 (16 rows), col range (w&3)*128 (8 N-tiles).
// ---------------------------------------------------------------------------
__launch_bounds__(256)
__global__ void attn_scores(const float* __restrict__ V,
                            const _Float16* __restrict__ UwT,
                            const float* __restrict__ whUb,
                            const float* __restrict__ vw,
                            const float* __restrict__ vb,
                            float* __restrict__ escore) {
    __shared__ __align__(16) _Float16 ldsA[32 * A_STRIDE];
    __shared__ __align__(16) _Float16 ldsB[512 * B_STRIDE];
    __shared__ float ldsVw[ATT];
    __shared__ float ldsE[32];

    const int tid     = threadIdx.x;
    const int lane    = tid & 31;
    const int wave    = tid >> 5;            // 0..7
    const int sm      = wave >> 2;           // 0..1  (row subtile)
    const int colbase = (wave & 3) * 128;    // col range of this wave
    const int rowbase = blockIdx.x * 32;

    ldsVw[tid]       = vw[tid];
    ldsVw[tid + 256] = vw[tid + 256];
    if (tid < 32) ldsE[tid] = 0.f;

    v8f acc[8];
#pragma unroll
    for (int t = 0; t < 8; ++t)
#pragma unroll
        for (int e = 0; e < 8; ++e) acc[t][e] = 0.f;

    // A loader assignment: thread -> (row = tid>>3, 4 consecutive K)
    const int arow  = tid >> 3;          // 0..31
    const int akoff = (tid & 7) * 4;     // 0..28
    const float* aGlob = V + (size_t)(rowbase + arow) * VDIM + akoff;
    _Float16* aLds = ldsA + arow * A_STRIDE + akoff;

    const int n15 = lane & 15;
    const int hi  = lane >> 4;           // 0 or 1 (half-wave)

    for (int kc = 0; kc < VDIM; kc += 32) {
        __syncthreads();  // LDS (re)fill safe; also covers ldsVw/ldsE init

        // ---- stage B tile: 512 cols x 32 K f16 (col-major K-contiguous) ----
#if defined(HAVE_ASYNC_LDS)
#pragma unroll
        for (int i = 0; i < 8; ++i) {
            const int u = tid + i * 256;     // 16-byte unit id, 2048 units
            __builtin_amdgcn_global_load_async_to_lds_b128(
                (glob_v4i*)(UwT + (size_t)(u >> 2) * VDIM + kc + (u & 3) * 8),
                (lds_v4i*)(ldsB + (u >> 2) * B_STRIDE + (u & 3) * 8),
                /*offset=*/0, /*cpol=*/0);
        }
#else
        uint4 tmpB[8];
#pragma unroll
        for (int i = 0; i < 8; ++i) {
            const int u = tid + i * 256;
            tmpB[i] = *(const uint4*)(UwT + (size_t)(u >> 2) * VDIM + kc +
                                      (u & 3) * 8);
        }
#endif
        // ---- stage A tile: 32 rows x 32 K, fp32 -> f16 in registers ----
        v4f fa = *(const v4f*)(aGlob + kc);
        v4h ha;
#pragma unroll
        for (int j = 0; j < 4; ++j) ha[j] = (_Float16)fa[j];
        *(v4h*)aLds = ha;

#if defined(HAVE_ASYNC_LDS)
        __builtin_amdgcn_s_wait_asynccnt(0);
#else
#pragma unroll
        for (int i = 0; i < 8; ++i) {
            const int u = tid + i * 256;
            *(uint4*)(ldsB + (u >> 2) * B_STRIDE + (u & 3) * 8) = tmpB[i];
        }
#endif
        __syncthreads();

        // ---- WMMA: A frag per ISA 16-bit 16x32 layout ----
        const int aoff = (sm * 16 + n15) * A_STRIDE + hi * 8;
        v8h a0 = *(const v8h*)(ldsA + aoff);        // K {0..7}/{8..15}
        v8h a1 = *(const v8h*)(ldsA + aoff + 16);   // K {16..23}/{24..31}
        v16h afrag = __builtin_shufflevector(a0, a1, 0, 1, 2, 3, 4, 5, 6, 7,
                                             8, 9, 10, 11, 12, 13, 14, 15);

        // B frags pipelined one N-tile ahead so wmma overlaps ds_loads.
        const int bbase = (colbase + n15) * B_STRIDE + hi * 8;
        v8h b0 = *(const v8h*)(ldsB + bbase);
        v8h b1 = *(const v8h*)(ldsB + bbase + 16);
#pragma unroll
        for (int nt = 0; nt < 8; ++nt) {
            v16h bfrag = __builtin_shufflevector(b0, b1, 0, 1, 2, 3, 4, 5, 6,
                                                 7, 8, 9, 10, 11, 12, 13, 14,
                                                 15);
            if (nt < 7) {
                const int boff = bbase + (nt + 1) * 16 * B_STRIDE;
                b0 = *(const v8h*)(ldsB + boff);
                b1 = *(const v8h*)(ldsB + boff + 16);
            }
            acc[nt] = __builtin_amdgcn_wmma_f32_16x16x32_f16(
                false, afrag, false, bfrag, (short)0, acc[nt], false, false);
        }
    }
    __syncthreads();

    // ---- epilogue: e partial = sum_a tanh(acc + whUb) * vw[a] per row ----
    // C/D layout: VGPR i -> M = i + 8*(lane>=16), N = lane&15
#pragma unroll
    for (int i = 0; i < 8; ++i) {
        const int lrow = sm * 16 + hi * 8 + i;
        const int r    = rowbase + lrow;
        const int bidx = r / N_SZ;
        const float* whp = whUb + bidx * ATT;
        float s = 0.f;
#pragma unroll
        for (int nt = 0; nt < 8; ++nt) {
            const int a   = colbase + nt * 16 + n15;
            const float p = acc[nt][i] + whp[a];
            s += fast_tanh(p) * ldsVw[a];
        }
        atomicAdd(&ldsE[lrow], s);
    }
    __syncthreads();
    if (tid < 32) escore[rowbase + tid] = ldsE[tid] + vb[0];
}

// ---------------------------------------------------------------------------
// Kernel 4: softmax over N=196 per batch, writes attention weights output.
// ---------------------------------------------------------------------------
__global__ void softmax_kernel(const float* __restrict__ escore,
                               float* __restrict__ aout) {
    __shared__ float red[256];
    const int b = blockIdx.x;
    const int t = threadIdx.x;
    const float v = (t < N_SZ) ? escore[b * N_SZ + t] : -__builtin_inff();
    red[t] = v;
    __syncthreads();
    for (int s = 128; s > 0; s >>= 1) {
        if (t < s) red[t] = fmaxf(red[t], red[t + s]);
        __syncthreads();
    }
    const float mx = red[0];
    __syncthreads();
    const float e = (t < N_SZ) ? fast_exp(v - mx) : 0.f;
    red[t] = e;
    __syncthreads();
    for (int s = 128; s > 0; s >>= 1) {
        if (t < s) red[t] += red[t + s];
        __syncthreads();
    }
    const float inv = 1.f / red[0];
    if (t < N_SZ) aout[b * N_SZ + t] = e * inv;
}

// ---------------------------------------------------------------------------
// Kernel 5: ctx[b,v] = sum_n a[b,n] * V[b,n,v]   (HBM-bound, coalesced)
// ---------------------------------------------------------------------------
__global__ void ctx_kernel(const float* __restrict__ V,
                           const float* __restrict__ a,
                           float* __restrict__ ctx) {
    const int b = blockIdx.y;
    const int v = blockIdx.x * 256 + threadIdx.x;
    const float* Vb = V + (size_t)b * N_SZ * VDIM + v;
    const float* ab = a + b * N_SZ;
    float acc = 0.f;
    for (int n = 0; n < N_SZ; ++n) acc += ab[n] * Vb[(size_t)n * VDIM];
    ctx[(size_t)b * VDIM + v] = acc;
}

// ---------------------------------------------------------------------------
extern "C" void kernel_launch(void* const* d_in, const int* in_sizes, int n_in,
                              void* d_out, int out_size, void* d_ws,
                              size_t ws_size, hipStream_t stream) {
    const float* h  = (const float*)d_in[0];
    const float* V  = (const float*)d_in[1];
    const float* Ww = (const float*)d_in[2];
    const float* Wb = (const float*)d_in[3];
    const float* Uw = (const float*)d_in[4];
    const float* Ub = (const float*)d_in[5];
    const float* vw = (const float*)d_in[6];
    const float* vb = (const float*)d_in[7];

    float* out  = (float*)d_out;
    float* ctx  = out;                       // [128][2048]
    float* attn = out + (size_t)B_SZ * VDIM; // [128][196]

    char* ws = (char*)d_ws;
    _Float16* UwT = (_Float16*)ws;                                   // 2 MB
    float* whUb   = (float*)(ws + (size_t)ATT * VDIM * 2);           // 256 KB
    float* escore = (float*)(ws + (size_t)ATT * VDIM * 2 +
                             (size_t)B_SZ * ATT * 4);                // 100 KB

    uw_transpose_f16<<<dim3(VDIM / 32, ATT / 32), 256, 0, stream>>>(Uw, UwT);
    wh_kernel<<<dim3(ATT / 256, B_SZ), 256, 0, stream>>>(h, Ww, Wb, Ub, whUb);
    attn_scores<<<dim3(ROWS / 32), 256, 0, stream>>>(V, UwT, whUb, vw, vb,
                                                     escore);
    softmax_kernel<<<dim3(B_SZ), 256, 0, stream>>>(escore, attn);
    ctx_kernel<<<dim3(VDIM / 256, B_SZ), 256, 0, stream>>>(V, attn, ctx);
}
